// PrototypeLayer_69913477644512
// MI455X (gfx1250) — matrix operand order repaired and checked by reading, not compiled, not run on hardware
//
#include <hip/hip_runtime.h>
#include <hip/hip_bf16.h>

// PrototypeLayer on MI455X (gfx1250):
//   xp = einsum('bchw,pc->bphw'); dist = max(x2 - 2xp + p2, 0); log((dist+1)/(dist+eps))
// GEMM M=2000 protos x N=12544 pixels x K=512 as split-bf16 (hi+lo) on the XDL
// path: 3x v_wmma_f32_16x16x32_bf16 per K=32, f32 accumulation (~fp32 accuracy).
// Precompute (once, into d_ws ~28.5 MB): bf16 hi/lo planes of prototypes
// (k-contiguous) and of x TRANSPOSED to pixel-major [12544][512] so GEMM staging
// is pure b128 traffic; plus |x|^2 per pixel and |p|^2 per prototype.

typedef __attribute__((ext_vector_type(16))) __bf16 v16bf;
typedef __attribute__((ext_vector_type(8)))  __bf16 v8bf;
typedef __attribute__((ext_vector_type(4)))  __bf16 v4bf;
typedef __attribute__((ext_vector_type(8)))  float  v8f;

#define CH        512
#define HW        196
#define NBATCH    64
#define NPIX      (NBATCH * HW)      // 12544
#define NPROT     2000
#define XBSTRIDE  (CH * HW)          // 100352 floats per batch image
#define OBSTRIDE  (NPROT * HW)       // 392000 floats per batch output

// ---------------------------------------------------------------- precompute
// Prototypes: elementwise fp32 -> (bf16 hi, bf16 lo), float4-vectorized.
__global__ __launch_bounds__(256) void split_bf16_kernel(
    const float* __restrict__ in, __bf16* __restrict__ hi,
    __bf16* __restrict__ lo, int n4) {
  int i = blockIdx.x * 256 + threadIdx.x;
  if (i >= n4) return;
  float4 v = reinterpret_cast<const float4*>(in)[i];
  __bf16 h0 = (__bf16)v.x, h1 = (__bf16)v.y, h2 = (__bf16)v.z, h3 = (__bf16)v.w;
  __bf16 l0 = (__bf16)(v.x - (float)h0);
  __bf16 l1 = (__bf16)(v.y - (float)h1);
  __bf16 l2 = (__bf16)(v.z - (float)h2);
  __bf16 l3 = (__bf16)(v.w - (float)h3);
  v4bf hv = {h0, h1, h2, h3};
  v4bf lv = {l0, l1, l2, l3};
  reinterpret_cast<v4bf*>(hi)[i] = hv;
  reinterpret_cast<v4bf*>(lo)[i] = lv;
}

// x: fp32 (64,512,14,14) -> bf16 hi/lo planes transposed to [m][c] (m=b*196+hw).
// grid = (49, 32): blockIdx.x*256+tid = pixel m (reads coalesced along hw),
// blockIdx.y picks a 16-channel slab; each thread writes 2x16 B per plane.
__global__ __launch_bounds__(256) void splitT_x_kernel(
    const float* __restrict__ x, __bf16* __restrict__ hi,
    __bf16* __restrict__ lo) {
  const int m  = blockIdx.x * 256 + threadIdx.x;
  const int c0 = blockIdx.y * 16;
  const int b  = m / HW;
  const int hw = m - b * HW;
  const float* px = x + b * XBSTRIDE + c0 * HW + hw;
  __bf16 h[16], l[16];
#pragma unroll
  for (int i = 0; i < 16; ++i) {
    float v = px[i * HW];
    h[i] = (__bf16)v;
    l[i] = (__bf16)(v - (float)h[i]);
  }
  v8bf hv0 = {h[0], h[1], h[2],  h[3],  h[4],  h[5],  h[6],  h[7]};
  v8bf hv1 = {h[8], h[9], h[10], h[11], h[12], h[13], h[14], h[15]};
  v8bf lv0 = {l[0], l[1], l[2],  l[3],  l[4],  l[5],  l[6],  l[7]};
  v8bf lv1 = {l[8], l[9], l[10], l[11], l[12], l[13], l[14], l[15]};
  __bf16* ph = hi + m * CH + c0;
  __bf16* pl = lo + m * CH + c0;
  *reinterpret_cast<v8bf*>(ph)     = hv0;
  *reinterpret_cast<v8bf*>(ph + 8) = hv1;
  *reinterpret_cast<v8bf*>(pl)     = lv0;
  *reinterpret_cast<v8bf*>(pl + 8) = lv1;
}

// x2[m] = sum_c x[b,c,hw]^2, coalesced along hw.
__global__ __launch_bounds__(256) void x2_kernel(
    const float* __restrict__ x, float* __restrict__ x2) {
  int m = blockIdx.x * 256 + threadIdx.x;
  if (m >= NPIX) return;
  int b = m / HW, hw = m - b * HW;
  const float* px = x + b * XBSTRIDE + hw;
  float s = 0.0f;
#pragma unroll 8
  for (int c = 0; c < CH; ++c) { float v = px[c * HW]; s = fmaf(v, v, s); }
  x2[m] = s;
}

// p2[n] = sum_c p[n,c]^2, contiguous float4 reads.
__global__ __launch_bounds__(256) void p2_kernel(
    const float* __restrict__ p, float* __restrict__ p2) {
  int n = blockIdx.x * 256 + threadIdx.x;
  if (n >= NPROT) return;
  const float4* pp = reinterpret_cast<const float4*>(p + n * CH);
  float s = 0.0f;
#pragma unroll 4
  for (int i = 0; i < CH / 4; ++i) {
    float4 v = pp[i];
    s = fmaf(v.x, v.x, s); s = fmaf(v.y, v.y, s);
    s = fmaf(v.z, v.z, s); s = fmaf(v.w, v.w, s);
  }
  p2[n] = s;
}

// ---------------------------------------------------------------- main GEMM
// Block: 64 protos x 128 pixels, 4 waves (2x2), each wave 32x64 = 2x4 WMMA
// tiles -> 24 WMMA per 12 fragment loads per K=32 chunk. Double-buffered LDS,
// one barrier per chunk. Row stride 40 bf16 (80 B): lane l hits bank 20*l
// mod 64 -> conflict-free ds_load_b128, 16 B aligned.
#define BP   64
#define BX   128
#define KC   32
#define NKC  (CH / KC)   // 16
#define LST  40          // LDS row stride in bf16 elements

static __device__ __forceinline__ v16bf join8(v8bf a, v8bf b) {
  return __builtin_shufflevector(a, b, 0, 1, 2, 3, 4, 5, 6, 7,
                                        8, 9, 10, 11, 12, 13, 14, 15);
}

__global__ __launch_bounds__(128) void proto_gemm_kernel(
    const __bf16* __restrict__ xhi, const __bf16* __restrict__ xlo,
    const __bf16* __restrict__ phi, const __bf16* __restrict__ plo,
    const float* __restrict__ x2w, const float* __restrict__ p2w,
    float* __restrict__ out) {
  __shared__ __bf16 Ah[2][BP * LST];   // prototypes hi (WMMA M)
  __shared__ __bf16 Al[2][BP * LST];   // prototypes lo
  __shared__ __bf16 Bh[2][BX * LST];   // pixels hi     (WMMA N)
  __shared__ __bf16 Bl[2][BX * LST];   // pixels lo

  const int tid  = threadIdx.x;
  const int lane = tid & 31;
  const int wave = tid >> 5;
  const int wrow = wave >> 1;          // proto half (32 rows each)
  const int wcol = wave & 1;           // pixel half (64 cols each)
  const int p0 = blockIdx.x * BP;
  const int m0 = blockIdx.y * BX;

  // A staging: 2 threads per proto row, 16 contiguous k (32 B) each.
  const int ap  = tid >> 1;
  const int akh = (tid & 1) * 16;
  int apg = p0 + ap; if (apg > NPROT - 1) apg = NPROT - 1;   // pad clamp
  const __bf16* aH = phi + apg * CH + akh;
  const __bf16* aL = plo + apg * CH + akh;

  // B staging: 1 thread per pixel row, 32 contiguous k (64 B) each.
  const __bf16* bH = xhi + (m0 + tid) * CH;
  const __bf16* bL = xlo + (m0 + tid) * CH;

  v8f acc[2][4];
#pragma unroll
  for (int i = 0; i < 2; ++i)
#pragma unroll
    for (int j = 0; j < 4; ++j)
      acc[i][j] = (v8f){0.f, 0.f, 0.f, 0.f, 0.f, 0.f, 0.f, 0.f};

  const int half = lane >> 4;
  const int rsel = lane & 15;
  const int aoff = ap * LST + akh;

  v8bf ra0, ra1, ra2, ra3, rb[8];

#define FETCH(kb)                                                     \
  ra0 = *reinterpret_cast<const v8bf*>(aH + (kb));                    \
  ra1 = *reinterpret_cast<const v8bf*>(aH + (kb) + 8);                \
  ra2 = *reinterpret_cast<const v8bf*>(aL + (kb));                    \
  ra3 = *reinterpret_cast<const v8bf*>(aL + (kb) + 8);                \
  _Pragma("unroll")                                                   \
  for (int q = 0; q < 4; ++q) {                                       \
    rb[q]     = *reinterpret_cast<const v8bf*>(bH + (kb) + q * 8);    \
    rb[q + 4] = *reinterpret_cast<const v8bf*>(bL + (kb) + q * 8);    \
  }

#define STAGE(buf)                                                    \
  *reinterpret_cast<v8bf*>(&Ah[buf][aoff])     = ra0;                 \
  *reinterpret_cast<v8bf*>(&Ah[buf][aoff + 8]) = ra1;                 \
  *reinterpret_cast<v8bf*>(&Al[buf][aoff])     = ra2;                 \
  *reinterpret_cast<v8bf*>(&Al[buf][aoff + 8]) = ra3;                 \
  _Pragma("unroll")                                                   \
  for (int q = 0; q < 4; ++q) {                                       \
    *reinterpret_cast<v8bf*>(&Bh[buf][tid * LST + q * 8]) = rb[q];    \
    *reinterpret_cast<v8bf*>(&Bl[buf][tid * LST + q * 8]) = rb[q + 4];\
  }

  FETCH(0)
  STAGE(0)
  __syncthreads();

  for (int c = 0; c < NKC; ++c) {
    const int cur = c & 1;
    if (c + 1 < NKC) { FETCH((c + 1) * KC) }   // overlap with WMMAs below

    // A fragments: lanes 0-15 need K 0-7 & 16-23, lanes 16-31 K 8-15 & 24-31.
    v16bf afh[2], afl[2];
#pragma unroll
    for (int t = 0; t < 2; ++t) {
      const int ar = (wrow * 32 + t * 16 + rsel) * LST;
      afh[t] = join8(*reinterpret_cast<const v8bf*>(&Ah[cur][ar + half * 8]),
                     *reinterpret_cast<const v8bf*>(&Ah[cur][ar + 16 + half * 8]));
      afl[t] = join8(*reinterpret_cast<const v8bf*>(&Al[cur][ar + half * 8]),
                     *reinterpret_cast<const v8bf*>(&Al[cur][ar + 16 + half * 8]));
    }
    // B fragments: lanes 0-15 hold K 0-7 & 16-23, lanes 16-31 K 8-15 & 24-31.
#pragma unroll
    for (int j = 0; j < 4; ++j) {
      const int br = (wcol * 64 + j * 16 + rsel) * LST;
      v16bf bfh = join8(*reinterpret_cast<const v8bf*>(&Bh[cur][br + half * 8]),
                        *reinterpret_cast<const v8bf*>(&Bh[cur][br + 16 + half * 8]));
      v16bf bfl = join8(*reinterpret_cast<const v8bf*>(&Bl[cur][br + half * 8]),
                        *reinterpret_cast<const v8bf*>(&Bl[cur][br + 16 + half * 8]));
#pragma unroll
      for (int i = 0; i < 2; ++i) {
        acc[i][j] = __builtin_amdgcn_wmma_f32_16x16x32_bf16(
            false, afh[i], false, bfh, (short)0, acc[i][j], false, false);
        acc[i][j] = __builtin_amdgcn_wmma_f32_16x16x32_bf16(
            false, afh[i], false, bfl, (short)0, acc[i][j], false, false);
        acc[i][j] = __builtin_amdgcn_wmma_f32_16x16x32_bf16(
            false, afl[i], false, bfh, (short)0, acc[i][j], false, false);
      }
    }

    if (c + 1 < NKC) { STAGE(cur ^ 1) }
    __syncthreads();
  }
#undef FETCH
#undef STAGE

  // Epilogue: dist = max(x2 - 2xp + p2, 0); log((dist+1)/(dist+eps)).
  // C layout: VGPR r -> proto row r (+8 for lanes 16-31); lane%16 -> pixel.
  const int h8 = half * 8;
#pragma unroll
  for (int j = 0; j < 4; ++j) {
    const int m  = m0 + wcol * 64 + j * 16 + rsel;
    const int b  = m / HW;
    const int hw = m - b * HW;
    const float x2v = x2w[m];
    float* ob = out + b * OBSTRIDE + hw;
#pragma unroll
    for (int i = 0; i < 2; ++i) {
      const int pb = p0 + wrow * 32 + i * 16 + h8;
#pragma unroll
      for (int r = 0; r < 8; ++r) {
        const int p = pb + r;
        if (p < NPROT) {
          float xp = acc[i][j][r];
          float d  = fmaxf(fmaf(-2.0f, xp, x2v + p2w[p]), 0.0f);
          ob[p * HW] = logf((d + 1.0f) / (d + 1e-4f));
        }
      }
    }
  }
}

// ---------------------------------------------------------------- launch
extern "C" void kernel_launch(void* const* d_in, const int* in_sizes, int n_in,
                              void* d_out, int out_size, void* d_ws, size_t ws_size,
                              hipStream_t stream) {
  const float* x     = (const float*)d_in[0];   // (64,512,14,14)
  const float* proto = (const float*)d_in[1];   // (2000,512,1,1)
  float* out = (float*)d_out;                   // (64,2000,14,14)

  const int NX = NPIX * CH;                     // 6,422,528
  const int NP = NPROT * CH;                    // 1,024,000

  __bf16* xhi = (__bf16*)d_ws;                  // [12544][512] pixel-major
  __bf16* xlo = xhi + NX;
  __bf16* phi = xlo + NX;                       // [2000][512]
  __bf16* plo = phi + NP;
  float*  x2w = (float*)(plo + NP);
  float*  p2w = x2w + NPIX;
  // ws usage: 4*NX + 4*NP + 4*(NPIX+NPROT) bytes ~= 28.5 MB

  splitT_x_kernel<<<dim3(NPIX / 256, CH / 16), 256, 0, stream>>>(x, xhi, xlo);
  split_bf16_kernel<<<NP / 4 / 256, 256, 0, stream>>>(proto, phi, plo, NP / 4);
  x2_kernel<<<(NPIX + 255) / 256, 256, 0, stream>>>(x, x2w);
  p2_kernel<<<(NPROT + 255) / 256, 256, 0, stream>>>(proto, p2w);

  dim3 grid((NPROT + BP - 1) / BP, NPIX / BX);  // 32 x 98
  proto_gemm_kernel<<<grid, 128, 0, stream>>>(xhi, xlo, phi, plo, x2w, p2w, out);
}
